// MambaKTHeadModel_24412594110752
// MI455X (gfx1250) — matrix operand order repaired and checked by reading, not compiled, not run on hardware
//
#include <hip/hip_runtime.h>
#include <cstdint>
#include <cstddef>

// ---------------------------------------------------------------------------
// MambaKT head model for MI455X (gfx1250, wave32, WMMA bf16 16x16x32).
// All heavy GEMMs run through v_wmma_f32_16x16x32_bf16 with f32 accumulate.
// B operands are pre-packed into WMMA fragment order so each lane loads its
// whole 16-element fragment as one contiguous 32-byte vector load.
// ---------------------------------------------------------------------------

typedef __bf16 bf16;
typedef __attribute__((ext_vector_type(16))) bf16  v16bf;
typedef __attribute__((ext_vector_type(8)))  bf16  v8bf;
typedef __attribute__((ext_vector_type(8)))  float v8f;

#define DEVFN __device__ __forceinline__

DEVFN float rcpfast(float x) { return __builtin_amdgcn_rcpf(x); }
DEVFN float sigf(float x) { return rcpfast(1.f + __expf(-x)); }
DEVFN float siluf(float x) { return x * sigf(x); }

DEVFN v16bf load_afrag(const bf16* Arow, int k0) {
    v8bf lo = *(const v8bf*)(Arow + k0);        // k = k0+hf*8 .. +7
    v8bf hi = *(const v8bf*)(Arow + k0 + 16);   // k = k0+hf*8+16 .. +23
    return __builtin_shufflevector(lo, hi, 0, 1, 2, 3, 4, 5, 6, 7,
                                   8, 9, 10, 11, 12, 13, 14, 15);
}

// ---------------- WMMA GEMM --------------------------------------------------
// C[M,N] = act(A[M,K]bf16(row) @ Bpacked[K,N]bf16 + bias).  M%16==0, N%16==0,
// K%32==0.  Each wave32 computes a 16x32 strip (two 16x16 tiles share A);
// an odd trailing N-tile is handled by a separate single-tile loop so the
// steady-state K-loop has no EXEC manipulation around the WMMAs.
// Bpacked layout: [K/32][N/16][lane 0..31][e 0..15], where
//   k = kb*32 + (lane>>4)*8 + (e&7) + (e>>3)*16,  n = nt*16 + (lane&15).
// act: 0=none 1=relu 2=silu 3=sigmoid
__global__ void __launch_bounds__(128)
k_gemm_bf16(const bf16* __restrict__ A, int lda,
            const bf16* __restrict__ Bp,
            float* __restrict__ C, int ldc,
            const float* __restrict__ bias,
            int M, int N, int K, int act)
{
    const int lane   = threadIdx.x;            // 0..31
    const int ntiles = N >> 4;
    const int npairs = (ntiles + 1) >> 1;
    const int work   = (M >> 4) * npairs;
    const int w      = blockIdx.x * blockDim.y + threadIdx.y;
    if (w >= work) return;
    const int  m0   = (w / npairs) << 4;
    const int  nt0  = (w % npairs) << 1;
    const bool has2 = (nt0 + 1) < ntiles;      // wave-uniform, loop-invariant
    const int  r    = lane & 15;
    const int  hf   = lane >> 4;

    v8f acc0 = {}, acc1 = {};
    const bf16* Arow  = A + (size_t)(m0 + r) * lda + hf * 8;
    const bf16* Bbase = Bp + (size_t)lane * 16 + (size_t)nt0 * 512;

    if (has2) {
        for (int k0 = 0; k0 < K; k0 += 32) {
            v16bf a = load_afrag(Arow, k0);
            const bf16* Bk = Bbase + (size_t)(k0 >> 5) * ntiles * 512;
            v16bf b0 = *(const v16bf*)(Bk);
            v16bf b1 = *(const v16bf*)(Bk + 512);
            acc0 = __builtin_amdgcn_wmma_f32_16x16x32_bf16(
                false, a, false, b0, (short)0, acc0, false, false);
            acc1 = __builtin_amdgcn_wmma_f32_16x16x32_bf16(
                false, a, false, b1, (short)0, acc1, false, false);
        }
    } else {
        for (int k0 = 0; k0 < K; k0 += 32) {
            v16bf a = load_afrag(Arow, k0);
            const bf16* Bk = Bbase + (size_t)(k0 >> 5) * ntiles * 512;
            v16bf b0 = *(const v16bf*)(Bk);
            acc0 = __builtin_amdgcn_wmma_f32_16x16x32_bf16(
                false, a, false, b0, (short)0, acc0, false, false);
        }
    }

    // epilogue
    for (int t = 0; t < 2; ++t) {
        if (t == 1 && !has2) break;
        v8f acc = (t == 0) ? acc0 : acc1;
        const int   n  = ((nt0 + t) << 4) + r;
        const float bv = bias ? bias[n] : 0.f;
#pragma unroll
        for (int rr = 0; rr < 8; ++rr) {
            int   m = m0 + rr + hf * 8;
            float v = acc[rr] + bv;
            if (act == 1)      v = fmaxf(v, 0.f);
            else if (act == 2) v = siluf(v);
            else if (act == 3) v = sigf(v);
            C[(size_t)m * ldc + n] = v;
        }
    }
}

// ---------------- pack B into WMMA fragment order ---------------------------
// transpose==0: B[k][n] = src[k*N + n]   (src is K x N row-major)
// transpose==1: B[k][n] = src[n*K + k]   (src is N x K row-major, i.e. W)
__global__ void k_packB(const float* __restrict__ src, bf16* __restrict__ dst,
                        int K, int N, int transpose)
{
    size_t i = (size_t)blockIdx.x * blockDim.x + threadIdx.x;
    if (i >= (size_t)K * N) return;
    const int ntiles = N >> 4;
    const int per_kb = ntiles * 512;
    int kb  = (int)(i / per_kb);
    int rem = (int)(i % per_kb);
    int nt  = rem / 512;
    int r2  = rem % 512;
    int lane = r2 >> 4, e = r2 & 15;
    int k = kb * 32 + (lane >> 4) * 8 + (e & 7) + (e >> 3) * 16;
    int n = nt * 16 + (lane & 15);
    float v = transpose ? src[(size_t)n * K + k] : src[(size_t)k * N + n];
    dst[i] = (bf16)v;
}

// ---------------- utility kernels ------------------------------------------
__global__ void k_cvt(const float* __restrict__ s, bf16* __restrict__ d, size_t n) {
    size_t i = (size_t)blockIdx.x * blockDim.x + threadIdx.x;
    if (i < n) d[i] = (bf16)s[i];
}
__global__ void k_fill(float* p, size_t n, float v) {
    size_t i = (size_t)blockIdx.x * blockDim.x + threadIdx.x;
    if (i < n) p[i] = v;
}
__global__ void k_copy(float* __restrict__ d, const float* __restrict__ s, size_t n) {
    size_t i = (size_t)blockIdx.x * blockDim.x + threadIdx.x;
    if (i < n) d[i] = s[i];
}
__global__ void k_addto(float* __restrict__ d, const float* __restrict__ s, size_t n) {
    size_t i = (size_t)blockIdx.x * blockDim.x + threadIdx.x;
    if (i < n) d[i] += s[i];
}

// ---------------- naive lin: O = act((in_relu?relu(A):A) @ W^T + bias) -----
// A: M x K (ld=lda), W: N x K (ld=ldw), O: M x N (ld=ldo). act 4 = softplus.
__global__ void k_lin(const float* __restrict__ A, int lda,
                      const float* __restrict__ W, int ldw,
                      const float* __restrict__ bias,
                      float* __restrict__ O, int ldo,
                      int M, int N, int K, int act, int in_relu)
{
    size_t i = (size_t)blockIdx.x * blockDim.x + threadIdx.x;
    if (i >= (size_t)M * N) return;
    int m = (int)(i / N), n = (int)(i % N);
    float s = bias ? bias[n] : 0.f;
    for (int k = 0; k < K; ++k) {
        float x = A[(size_t)m * lda + k];
        if (in_relu) x = fmaxf(x, 0.f);
        s += x * W[(size_t)n * ldw + k];
    }
    if (act == 1)      s = fmaxf(s, 0.f);
    else if (act == 2) s = siluf(s);
    else if (act == 3) s = sigf(s);
    else if (act == 4) s = (s > 20.f) ? s : log1pf(__expf(s));
    O[(size_t)m * ldo + n] = s;
}

// ---------------- model-specific kernels -----------------------------------
__global__ void k_stu_h(const float* __restrict__ semb, const int* __restrict__ student,
                        float* __restrict__ out) {
    int i = blockIdx.x * 256 + threadIdx.x;
    if (i >= 16 * 256) return;
    int b = i >> 8, f = i & 255;
    int s = student[b] - 1;
    out[i] = semb[(size_t)s * 256 + f];
}

__global__ void k_deg_edges(const int* __restrict__ skill, float* __restrict__ deg) {
    int i = blockIdx.x * 256 + threadIdx.x;
    if (i >= 16 * 499) return;
    int b = i / 499, j = i % 499;
    int t = skill[b * 500 + j + 1];
    atomicAdd(&deg[b * 513 + t], 1.f);
}

__global__ void k_gcn_init1(const float* __restrict__ hl1, const float* __restrict__ deg,
                            const float* __restrict__ b1, float* __restrict__ agg) {
    int i = blockIdx.x * 256 + threadIdx.x;
    if (i >= 16 * 513 * 8) return;
    int b = i / (513 * 8), rem = i % (513 * 8);
    int n = rem / 8, f = rem % 8;
    agg[i] = hl1[n * 8 + f] * rcpfast(deg[b * 513 + n]) + b1[f];
}
__global__ void k_gcn_edges1(const float* __restrict__ hl1, const float* __restrict__ deg,
                             const int* __restrict__ skill, float* __restrict__ agg) {
    int i = blockIdx.x * 256 + threadIdx.x;
    if (i >= 16 * 499 * 8) return;
    int b = i / (499 * 8), rem = i % (499 * 8);
    int j = rem / 8, f = rem % 8;
    int s = skill[b * 500 + j], t = skill[b * 500 + j + 1];
    float c = rsqrtf(deg[b * 513 + s]) * rsqrtf(deg[b * 513 + t]);
    atomicAdd(&agg[((size_t)b * 513 + t) * 8 + f], hl1[s * 8 + f] * c);
}
__global__ void k_gcn_init2(const float* __restrict__ hl2, const float* __restrict__ deg,
                            const float* __restrict__ b2, float* __restrict__ allh) {
    size_t i = (size_t)blockIdx.x * 256 + threadIdx.x;
    if (i >= (size_t)16 * 513 * 256) return;
    int b = (int)(i / (513 * 256)), rem = (int)(i % (513 * 256));
    int n = rem / 256, f = rem % 256;
    allh[i] = hl2[i] * rcpfast(deg[b * 513 + n]) + b2[f];
}
__global__ void k_gcn_edges2(const float* __restrict__ hl2, const float* __restrict__ deg,
                             const int* __restrict__ skill, float* __restrict__ allh) {
    size_t i = (size_t)blockIdx.x * 256 + threadIdx.x;
    if (i >= (size_t)16 * 499 * 256) return;
    int b = (int)(i / (499 * 256)), rem = (int)(i % (499 * 256));
    int j = rem / 256, f = rem % 256;
    int s = skill[b * 500 + j], t = skill[b * 500 + j + 1];
    float c = rsqrtf(deg[b * 513 + s]) * rsqrtf(deg[b * 513 + t]);
    atomicAdd(&allh[((size_t)b * 513 + t) * 256 + f],
              hl2[((size_t)b * 513 + s) * 256 + f] * c);
}

// pooled graph feature: softmax(pos[len-1]*mask) weighted sum of gathered rows
__global__ void k_graph_feat(const float* __restrict__ allh, const int* __restrict__ skill,
                             const int* __restrict__ answer, const float* __restrict__ pos,
                             float* __restrict__ gf) {
    __shared__ float sLog[500];
    __shared__ float sMask[500];
    __shared__ int   sIdx[500];
    __shared__ float red[256];
    int b = blockIdx.x, tid = threadIdx.x;

    float lensum = 0.f;
    for (int l = tid; l < 500; l += 256) {
        int a = answer[b * 500 + l];
        float mk = (a != 2) ? 1.f : 0.f;
        sMask[l] = mk;
        sIdx[l]  = skill[b * 500 + l];
        lensum += mk;
    }
    red[tid] = lensum; __syncthreads();
    for (int s = 128; s > 0; s >>= 1) { if (tid < s) red[tid] += red[tid + s]; __syncthreads(); }
    int li = (int)(red[0] + 0.5f) - 1; if (li < 0) li = 0;
    __syncthreads();

    float lmax = -1e30f;
    for (int l = tid; l < 500; l += 256) {
        float lg = pos[(size_t)li * 500 + l] * sMask[l];
        sLog[l] = lg;
        lmax = fmaxf(lmax, lg);
    }
    red[tid] = lmax; __syncthreads();
    for (int s = 128; s > 0; s >>= 1) { if (tid < s) red[tid] = fmaxf(red[tid], red[tid + s]); __syncthreads(); }
    float mx = red[0]; __syncthreads();

    float lsum = 0.f;
    for (int l = tid; l < 500; l += 256) lsum += __expf(sLog[l] - mx);
    red[tid] = lsum; __syncthreads();
    for (int s = 128; s > 0; s >>= 1) { if (tid < s) red[tid] += red[tid + s]; __syncthreads(); }
    float inv_denom = rcpfast(red[0]); __syncthreads();

    float acc = 0.f;   // tid == feature index (256 features)
    for (int l = 0; l < 500; ++l) {
        float w = __expf(sLog[l] - mx) * inv_denom * sMask[l];
        acc += allh[((size_t)b * 513 + sIdx[l]) * 256 + tid] * w;
    }
    gf[b * 256 + tid] = acc;
}

// build concat [skill_e | answer_e | seq] directly as bf16 rows of 768
__global__ void k_cat(const int* __restrict__ skill, const int* __restrict__ answer,
                      const float* __restrict__ skill_emb, const float* __restrict__ answer_emb,
                      const float* __restrict__ seq, bf16* __restrict__ cat) {
    size_t i = (size_t)blockIdx.x * 256 + threadIdx.x;
    if (i >= (size_t)8000 * 256) return;
    int row = (int)(i >> 8), f = (int)(i & 255);
    int b = row / 500, l = row % 500;
    size_t o = (size_t)row * 768;
    cat[o + f]       = (bf16)skill_emb[(size_t)skill[b * 500 + l] * 256 + f];
    cat[o + 256 + f] = (bf16)answer_emb[(size_t)answer[b * 500 + l] * 256 + f];
    cat[o + 512 + f] = (bf16)seq[b * 256 + f];
}

// layernorm over D=256, one row per block (256 threads)
__global__ void k_ln(const float* __restrict__ x, const float* __restrict__ w,
                     const float* __restrict__ b, float* __restrict__ o) {
    __shared__ float red[256];
    int row = blockIdx.x, tid = threadIdx.x;
    float v = x[(size_t)row * 256 + tid];
    red[tid] = v; __syncthreads();
    for (int s = 128; s > 0; s >>= 1) { if (tid < s) red[tid] += red[tid + s]; __syncthreads(); }
    float mean = red[0] * (1.f / 256.f); __syncthreads();
    float d = v - mean;
    red[tid] = d * d; __syncthreads();
    for (int s = 128; s > 0; s >>= 1) { if (tid < s) red[tid] += red[tid + s]; __syncthreads(); }
    float var = red[0] * (1.f / 256.f);
    o[(size_t)row * 256 + tid] = d * rsqrtf(var + 1e-5f) * w[tid] + b[tid];
}

// causal depthwise conv (DC=4) + SiLU on xc = xz[:, :512]; writes f32 + bf16
__global__ void k_conv_silu(const float* __restrict__ xz, const float* __restrict__ cw,
                            const float* __restrict__ cb, float* __restrict__ xc,
                            bf16* __restrict__ xcb) {
    size_t i = (size_t)blockIdx.x * 256 + threadIdx.x;
    if (i >= (size_t)8000 * 512) return;
    int row = (int)(i >> 9), c = (int)(i & 511);
    int b = row / 500, l = row % 500;
    float s = cb[c];
#pragma unroll
    for (int j = 0; j < 4; ++j) {
        int ll = l - 3 + j;
        if (ll >= 0) s += xz[((size_t)(b * 500 + ll)) * 1024 + c] * cw[c * 4 + j];
    }
    s = siluf(s);
    xc[i]  = s;
    xcb[i] = (bf16)s;
}

// selective scan: thread = channel e; 16 states in registers; B/C via LDS
__global__ void k_scan(const float* __restrict__ dt, const float* __restrict__ xdbl,
                       const float* __restrict__ xc, const float* __restrict__ xz,
                       const float* __restrict__ A_log, const float* __restrict__ Dp,
                       bf16* __restrict__ ybf) {
    __shared__ float sB[16], sC[16];
    int b = blockIdx.x >> 1, cb = blockIdx.x & 1;
    int e = cb * 256 + threadIdx.x;
    float A[16], h[16];
#pragma unroll
    for (int s = 0; s < 16; ++s) { A[s] = -__expf(A_log[(size_t)e * 16 + s]); h[s] = 0.f; }
    float dp = Dp[e];
    for (int l = 0; l < 500; ++l) {
        int row = b * 500 + l;
        if (threadIdx.x < 32) {
            float v = xdbl[(size_t)row * 48 + 16 + threadIdx.x];
            if (threadIdx.x < 16) sB[threadIdx.x] = v; else sC[threadIdx.x - 16] = v;
        }
        __syncthreads();
        float dtv = dt[(size_t)row * 512 + e];
        float xv  = xc[(size_t)row * 512 + e];
        float acc = 0.f;
#pragma unroll
        for (int s = 0; s < 16; ++s) {
            h[s] = h[s] * __expf(dtv * A[s]) + dtv * xv * sB[s];
            acc += h[s] * sC[s];
        }
        float zv = xz[(size_t)row * 1024 + 512 + e];
        float y  = (acc + xv * dp) * siluf(zv);
        ybf[(size_t)row * 512 + e] = (bf16)y;
        __syncthreads();
    }
}

__global__ void k_gate(const float* __restrict__ hD, const float* __restrict__ hH,
                       const float* __restrict__ gW, const float* __restrict__ gb,
                       float* __restrict__ gate) {
    int row = blockIdx.x * 256 + threadIdx.x;
    if (row >= 8000) return;
    float s = gb[0];
    for (int k = 0; k < 256; ++k)
        s += hD[(size_t)row * 256 + k] * gW[k] + hH[(size_t)row * 256 + k] * gW[256 + k];
    gate[row] = sigf(s);
}

__global__ void k_final(const float* __restrict__ lD, const float* __restrict__ lH,
                        const float* __restrict__ gate, const int* __restrict__ skill,
                        float* __restrict__ out) {
    int i = blockIdx.x * 256 + threadIdx.x;
    if (i >= 16 * 499) return;
    int b = i / 499, l = i % 499;
    int ns = skill[b * 500 + l + 1];
    float r = 0.f;
    if (ns < 512) {
        int row = b * 500 + l;
        float g  = gate[row];
        float pD = sigf(lD[(size_t)row * 512 + ns]);
        float pH = sigf(lH[(size_t)row * 512 + ns]);
        r = g * pD + (1.f - g) * pH;
    }
    out[i] = r;
}

// ---------------------------------------------------------------------------
extern "C" void kernel_launch(void* const* d_in, const int* in_sizes, int n_in,
                              void* d_out, int out_size, void* d_ws, size_t ws_size,
                              hipStream_t stream)
{
    (void)in_sizes; (void)n_in; (void)out_size; (void)ws_size;
    auto F = [&](int i) { return (const float*)d_in[i]; };
    auto I = [&](int i) { return (const int*)d_in[i]; };

    const int*   student   = I(0);
    const int*   skill     = I(1);
    const int*   answer    = I(2);
    const float* skill_emb = F(3);
    const float* answer_e  = F(4);
    const float* W_cdD = F(5);  const float* b_cdD = F(6);
    const float* W_cdH = F(7);  const float* b_cdH = F(8);
    const float* gcn1_W = F(9);  const float* gcn1_b = F(10);
    const float* gcn2_W = F(11); const float* gcn2_b = F(12);
    const float* stu_table = F(13);
    const float* G = F(14);
    const float* hg_W1 = F(15); const float* hg_b1 = F(16);
    const float* hg_W2 = F(17); const float* hg_b2 = F(18);
    const float* pos = F(19);
    const float* gate_W = F(20); const float* gate_b = F(21);
    const float* fcd_W = F(22);  const float* fcd_b = F(23);
    const float* fch_W = F(24);  const float* fch_b = F(25);
    // params_D (26..38) / params_H (39..51): pytree-sorted keys:
    // A_log, Dp, W_dt, W_in, W_out, W_x, b_dt, conv_b, conv_w, ln_b, ln_w, nf_b, nf_w

    char* base = (char*)d_ws;
    size_t cur = 0;
    auto alloc = [&](size_t bytes) -> void* {
        size_t a = (cur + 255) & ~(size_t)255;
        void* p = base + a;
        cur = a + bytes;
        return p;
    };
    auto aF = [&](size_t n) { return (float*)alloc(n * 4); };
    auto aB = [&](size_t n) { return (bf16*)alloc(n * 2); };

    auto gemm = [&](const bf16* A, int lda, const bf16* Bp, float* C, int ldc,
                    const float* bias, int M, int N, int K, int act) {
        int npairs = ((N / 16) + 1) / 2;
        int work = (M / 16) * npairs;
        k_gemm_bf16<<<dim3((work + 3) / 4), dim3(32, 4), 0, stream>>>(
            A, lda, Bp, C, ldc, bias, M, N, K, act);
    };
    auto cvt = [&](const float* s, bf16* d, size_t n) {
        k_cvt<<<dim3((unsigned)((n + 255) / 256)), 256, 0, stream>>>(s, d, n);
    };
    auto packB = [&](const float* s, bf16* d, int K, int N, int tr) {
        size_t n = (size_t)K * N;
        k_packB<<<dim3((unsigned)((n + 255) / 256)), 256, 0, stream>>>(s, d, K, N, tr);
    };
    auto grid1 = [&](size_t n) { return dim3((unsigned)((n + 255) / 256)); };

    // -------- persistent region --------
    float* stu_h   = aF(16 * 256);
    float* deg     = aF(16 * 513);
    float* all_h   = aF((size_t)16 * 513 * 256);
    float* gfeat   = aF(16 * 256);
    float* hD      = aF((size_t)8000 * 256);
    float* hH      = aF((size_t)8000 * 256);
    float* logitsD = aF((size_t)8000 * 512);
    float* logitsH = aF((size_t)8000 * 512);
    float* gate    = aF(8000);
    const size_t scratch0 = cur;

    // ================= phase 1: HGNN student embedding =================
    {
        cur = scratch0;
        bf16* Gb   = aB((size_t)4000 * 4000); cvt(G, Gb, (size_t)4000 * 4000);
        bf16* Sb   = aB((size_t)4000 * 256);  cvt(stu_table, Sb, (size_t)4000 * 256);
        bf16* W1p  = aB(256 * 256);           packB(hg_W1, W1p, 256, 256, 0);
        bf16* W2p  = aB(256 * 256);           packB(hg_W2, W2p, 256, 256, 0);
        float* t1  = aF((size_t)4000 * 256);
        gemm(Sb, 256, W1p, t1, 256, hg_b1, 4000, 256, 256, 0);
        bf16* t1p  = aB((size_t)4000 * 256);  packB(t1, t1p, 4000, 256, 0);
        float* x1  = aF((size_t)4000 * 256);
        gemm(Gb, 4000, t1p, x1, 256, nullptr, 4000, 256, 4000, 1);  // relu
        bf16* x1b  = aB((size_t)4000 * 256);  cvt(x1, x1b, (size_t)4000 * 256);
        float* t2  = aF((size_t)4000 * 256);
        gemm(x1b, 256, W2p, t2, 256, hg_b2, 4000, 256, 256, 0);
        bf16* t2p  = aB((size_t)4000 * 256);  packB(t2, t2p, 4000, 256, 0);
        float* semb = aF((size_t)4000 * 256);
        gemm(Gb, 4000, t2p, semb, 256, nullptr, 4000, 256, 4000, 0);
        k_stu_h<<<grid1(16 * 256), 256, 0, stream>>>(semb, student, stu_h);
    }

    // ================= phase 2: GCN + pooled graph feature =================
    {
        cur = scratch0;
        float* hl1 = aF(513 * 8);
        k_lin<<<grid1(513 * 8), 256, 0, stream>>>(skill_emb, 256, gcn1_W, 256, nullptr,
                                                  hl1, 8, 513, 8, 256, 0, 0);
        k_fill<<<grid1(16 * 513), 256, 0, stream>>>(deg, 16 * 513, 1.f);
        k_deg_edges<<<grid1(16 * 499), 256, 0, stream>>>(skill, deg);
        float* agg1 = aF((size_t)16 * 513 * 8);
        k_gcn_init1<<<grid1(16 * 513 * 8), 256, 0, stream>>>(hl1, deg, gcn1_b, agg1);
        k_gcn_edges1<<<grid1(16 * 499 * 8), 256, 0, stream>>>(hl1, deg, skill, agg1);
        float* hl2 = aF((size_t)16 * 513 * 256);
        k_lin<<<grid1((size_t)16 * 513 * 256), 256, 0, stream>>>(
            agg1, 8, gcn2_W, 8, nullptr, hl2, 256, 16 * 513, 256, 8, 0, /*in_relu*/1);
        k_gcn_init2<<<grid1((size_t)16 * 513 * 256), 256, 0, stream>>>(hl2, deg, gcn2_b, all_h);
        k_gcn_edges2<<<grid1((size_t)16 * 499 * 256), 256, 0, stream>>>(hl2, deg, skill, all_h);
        k_graph_feat<<<16, 256, 0, stream>>>(all_h, skill, answer, pos, gfeat);
    }

    // ================= phase 3: two Mamba backbones =================
    auto run_backbone = [&](const float* Wcd, const float* bcd, const float* seq,
                            int pb, float* hOut) {
        cur = scratch0;
        const float* A_log = F(pb + 0); const float* Dp    = F(pb + 1);
        const float* W_dt  = F(pb + 2); const float* W_in  = F(pb + 3);
        const float* W_out = F(pb + 4); const float* W_x   = F(pb + 5);
        const float* b_dt  = F(pb + 6); const float* convb = F(pb + 7);
        const float* convw = F(pb + 8); const float* ln_b  = F(pb + 9);
        const float* ln_w  = F(pb + 10); const float* nf_b = F(pb + 11);
        const float* nf_w  = F(pb + 12);

        bf16* catb = aB((size_t)8000 * 768);
        k_cat<<<grid1((size_t)8000 * 256), 256, 0, stream>>>(skill, answer, skill_emb,
                                                             answer_e, seq, catb);
        bf16* WcdP = aB((size_t)768 * 256); packB(Wcd, WcdP, 768, 256, 1);
        float* inX = aF((size_t)8000 * 256);
        gemm(catb, 768, WcdP, inX, 256, bcd, 8000, 256, 768, 0);

        float* resid = aF((size_t)8000 * 256);
        k_copy<<<grid1((size_t)8000 * 256), 256, 0, stream>>>(resid, inX, (size_t)8000 * 256);
        float* lnb  = aF((size_t)8000 * 256);
        float* hid  = aF((size_t)8000 * 256);
        bf16*  xbf  = aB((size_t)8000 * 256);
        bf16*  WinP = aB((size_t)256 * 1024);
        float* xz   = aF((size_t)8000 * 1024);
        float* xc   = aF((size_t)8000 * 512);
        bf16*  xcb  = aB((size_t)8000 * 512);
        bf16*  WxP  = aB((size_t)512 * 48);
        float* xdbl = aF((size_t)8000 * 48);
        float* dtb  = aF((size_t)8000 * 512);
        bf16*  ybf  = aB((size_t)8000 * 512);
        bf16*  WoutP = aB((size_t)512 * 256);

        for (int i = 0; i < 2; ++i) {
            if (i > 0)
                k_addto<<<grid1((size_t)8000 * 256), 256, 0, stream>>>(resid, hid, (size_t)8000 * 256);
            k_ln<<<8000, 256, 0, stream>>>(resid, ln_w + i * 256, ln_b + i * 256, lnb);
            cvt(lnb, xbf, (size_t)8000 * 256);
            packB(W_in + (size_t)i * 1024 * 256, WinP, 256, 1024, 1);
            gemm(xbf, 256, WinP, xz, 1024, nullptr, 8000, 1024, 256, 0);
            k_conv_silu<<<grid1((size_t)8000 * 512), 256, 0, stream>>>(
                xz, convw + (size_t)i * 512 * 4, convb + (size_t)i * 512, xc, xcb);
            packB(W_x + (size_t)i * 48 * 512, WxP, 512, 48, 1);
            gemm(xcb, 512, WxP, xdbl, 48, nullptr, 8000, 48, 512, 0);
            k_lin<<<grid1((size_t)8000 * 512), 256, 0, stream>>>(
                xdbl, 48, W_dt + (size_t)i * 512 * 16, 16, b_dt + (size_t)i * 512,
                dtb, 512, 8000, 512, 16, /*softplus*/4, 0);
            k_scan<<<32, 256, 0, stream>>>(dtb, xdbl, xc, xz,
                                           A_log + (size_t)i * 512 * 16,
                                           Dp + (size_t)i * 512, ybf);
            packB(W_out + (size_t)i * 256 * 512, WoutP, 512, 256, 1);
            gemm(ybf, 512, WoutP, hid, 256, nullptr, 8000, 256, 512, 0);
        }
        k_addto<<<grid1((size_t)8000 * 256), 256, 0, stream>>>(resid, hid, (size_t)8000 * 256);
        k_ln<<<8000, 256, 0, stream>>>(resid, nf_w, nf_b, hOut);
    };

    run_backbone(W_cdD, b_cdD, stu_h, 26, hD);
    run_backbone(W_cdH, b_cdH, gfeat, 39, hH);

    // ================= phase 4: heads + gate + pick =================
    {
        cur = scratch0;
        bf16* hDb = aB((size_t)8000 * 256); cvt(hD, hDb, (size_t)8000 * 256);
        bf16* hHb = aB((size_t)8000 * 256); cvt(hH, hHb, (size_t)8000 * 256);
        bf16* fcdP = aB((size_t)256 * 512); packB(fcd_W, fcdP, 256, 512, 1);
        bf16* fchP = aB((size_t)256 * 512); packB(fch_W, fchP, 256, 512, 1);
        gemm(hDb, 256, fcdP, logitsD, 512, fcd_b, 8000, 512, 256, 0);
        gemm(hHb, 256, fchP, logitsH, 512, fch_b, 8000, 512, 256, 0);
        k_gate<<<grid1(8000), 256, 0, stream>>>(hD, hH, gate_W, gate_b, gate);
        k_final<<<grid1(16 * 499), 256, 0, stream>>>(logitsD, logitsH, gate, skill,
                                                     (float*)d_out);
    }
}
